// Transformer_32676111188102
// MI455X (gfx1250) — matrix operand order repaired and checked by reading, not compiled, not run on hardware
//
#include <hip/hip_runtime.h>
#include <math.h>

// ---------------------------------------------------------------------------
// Transformer encoder-decoder forward for gfx1250 (MI455X).
// bf16 WMMA (v_wmma_f32_16x16x32_bf16) for all GEMMs + fused flash attention.
// f32 accumulate; softmax/LayerNorm in f32. Packed bf16 LDS staging +
// double-buffered GEMM mainloop + global_prefetch for the tile after next.
// ---------------------------------------------------------------------------

typedef __attribute__((ext_vector_type(16))) __bf16 v16bf;
typedef __attribute__((ext_vector_type(2)))  __bf16 bf16x2;
typedef __attribute__((ext_vector_type(8)))  float  v8f;

#define DEV_INLINE __device__ __forceinline__

static constexpr int BB  = 8;
static constexpr int SS  = 512;
static constexpr int DD  = 512;
static constexpr int HH  = 8;
static constexpr int DH  = 64;    // head dim
static constexpr int DFF = 2048;
static constexpr int NL  = 2;
static constexpr int NTOK = BB * SS; // 4096 rows in every GEMM

union V16U { uint4 q[2]; v16bf v; };

// Load a 16-element bf16 fragment as two 16-byte LDS reads (ds_load_b128).
DEV_INLINE v16bf ld_frag(const __bf16* p0, const __bf16* p1) {
  V16U u;
  u.q[0] = *(const uint4*)p0;
  u.q[1] = *(const uint4*)p1;
  return u.v;
}

// Two floats -> packed bf16 dword (v_cvt_pk_bf16_f32).
DEV_INLINE unsigned pack_bf16(float a, float b) {
#if __has_builtin(__builtin_amdgcn_cvt_pk_bf16_f32)
  union { bf16x2 v; unsigned u; } c;
  c.v = __builtin_amdgcn_cvt_pk_bf16_f32(a, b);
  return c.u;
#else
  union { __bf16 h[2]; unsigned u; } v;
  v.h[0] = (__bf16)a;
  v.h[1] = (__bf16)b;
  return v.u;
#endif
}

DEV_INLINE v8f wmma_bf16(v16bf a, v16bf b, v8f c) {
  return __builtin_amdgcn_wmma_f32_16x16x32_bf16(
      false, a, false, b, (short)0, c, false, false);
}

DEV_INLINE int imin(int a, int b) { return a < b ? a : b; }

// ---------------------------------------------------------------------------
// x = in + sinusoidal PE
// ---------------------------------------------------------------------------
__global__ __launch_bounds__(256) void add_pe_kernel(
    const float* __restrict__ in, float* __restrict__ out)
{
  int idx = blockIdx.x * 256 + threadIdx.x;
  int d = idx & (DD - 1);
  int s = (idx >> 9) & (SS - 1);
  int t = d >> 1;
  float div = expf((float)(2 * t) * (-logf(10000.f) / (float)DD));
  float ang = (float)s * div;
  float pe = (d & 1) ? cosf(ang) : sinf(ang);
  out[idx] = in[idx] + pe;
}

// ---------------------------------------------------------------------------
// GEMM: C[M=4096, N] = A[M, K] @ W[K, N] (+ bias) (opt. ReLU)
// BM=128, BN=128, BK=32; 256 threads = 8 waves, each wave a 32x64 tile
// (2x4 of 16x16 WMMA = 8 WMMA per k-step). Double-buffered LDS with
// register prefetch of tile s+1 and global_prefetch of tile s+2.
// ---------------------------------------------------------------------------
static constexpr int GBM = 128, GBN = 128, GBK = 32, GLD = GBK + 8; // 40

__global__ __launch_bounds__(256) void gemm_bf16_kernel(
    const float* __restrict__ A, const float* __restrict__ W,
    const float* __restrict__ bias, float* __restrict__ C,
    int N, int K, int relu)
{
  __shared__ __align__(16) __bf16 As[2][GBM][GLD];
  __shared__ __align__(16) __bf16 Bs[2][GBN][GLD]; // transposed: Bs[.][n][k]

  const int tid  = threadIdx.x;
  const int lane = tid & 31;
  const int ln   = lane & 15;
  const int lh   = lane >> 4;
  const int wave = tid >> 5;
  const int wm   = (wave & 3) * 32;
  const int wn   = (wave >> 2) * 64;
  const int bm   = blockIdx.y * GBM;
  const int bn   = blockIdx.x * GBN;

  const v8f vzero = {0.f,0.f,0.f,0.f,0.f,0.f,0.f,0.f};
  v8f acc[2][4] = {{vzero, vzero, vzero, vzero}, {vzero, vzero, vzero, vzero}};

  float4 ar[4];      // A prefetch: 4 x float4 (row, 4 consecutive k)
  float  br[8][2];   // B prefetch: 8 x (k, k+1) pairs at one column

  auto load_regs = [&](int k0) {
    #pragma unroll
    for (int i = 0; i < 4; ++i) {
      int chunk = i * 256 + tid;
      int r = chunk >> 3, c = (chunk & 7) * 4;
      ar[i] = *(const float4*)&A[(size_t)(bm + r) * K + k0 + c];
    }
    #pragma unroll
    for (int i = 0; i < 8; ++i) {
      int pidx = i * 256 + tid;
      int k2 = pidx >> 7, n = pidx & 127;
      const float* wp = &W[(size_t)(k0 + 2 * k2) * N + bn + n];
      br[i][0] = wp[0];
      br[i][1] = wp[N];
    }
  };
  auto store_lds = [&](int buf) {
    #pragma unroll
    for (int i = 0; i < 4; ++i) {
      int chunk = i * 256 + tid;
      int r = chunk >> 3, c = (chunk & 7) * 4;
      uint2 p;
      p.x = pack_bf16(ar[i].x, ar[i].y);
      p.y = pack_bf16(ar[i].z, ar[i].w);
      *(uint2*)&As[buf][r][c] = p;     // ds_store_b64
    }
    #pragma unroll
    for (int i = 0; i < 8; ++i) {
      int pidx = i * 256 + tid;
      int k2 = pidx >> 7, n = pidx & 127;
      *(unsigned*)&Bs[buf][n][2 * k2] = pack_bf16(br[i][0], br[i][1]); // b32
    }
  };
  // Hint HBM lines for the tile after next (global_prefetch_b8).
  auto prefetch_tile = [&](int k0) {
    const int r = tid & 127, half = tid >> 7;
    __builtin_prefetch(&A[(size_t)(bm + r) * K + k0 + half * 16], 0, 3);
    const int kk = tid >> 3, n16 = (tid & 7) * 16;
    __builtin_prefetch(&W[(size_t)(k0 + kk) * N + bn + n16], 0, 3);
  };

  load_regs(0);
  store_lds(0);
  __syncthreads();

  const int nsteps = K / GBK;
  int cur = 0;
  for (int s = 0; s < nsteps; ++s) {
    if (s + 1 < nsteps) load_regs((s + 1) * GBK);
    if (s + 2 < nsteps) prefetch_tile((s + 2) * GBK);

    v16bf af[2], bfr[4];
    #pragma unroll
    for (int sm = 0; sm < 2; ++sm) {
      const __bf16* row = &As[cur][wm + sm * 16 + ln][0];
      af[sm] = ld_frag(row + lh * 8, row + 16 + lh * 8);
    }
    #pragma unroll
    for (int sn = 0; sn < 4; ++sn) {
      const __bf16* row = &Bs[cur][wn + sn * 16 + ln][0];
      bfr[sn] = ld_frag(row + lh * 8, row + 16 + lh * 8);
    }
    #pragma unroll
    for (int sm = 0; sm < 2; ++sm)
      #pragma unroll
      for (int sn = 0; sn < 4; ++sn)
        acc[sm][sn] = wmma_bf16(af[sm], bfr[sn], acc[sm][sn]);

    if (s + 1 < nsteps) store_lds(cur ^ 1); // other buffer: no conflict with
                                            // this step's reads
    __syncthreads();
    cur ^= 1;
  }

  #pragma unroll
  for (int sm = 0; sm < 2; ++sm) {
    #pragma unroll
    for (int sn = 0; sn < 4; ++sn) {
      int col = bn + wn + sn * 16 + ln;
      float bv = bias ? bias[col] : 0.f;
      #pragma unroll
      for (int r = 0; r < 8; ++r) {
        int row = bm + wm + sm * 16 + r + 8 * lh;
        float v = acc[sm][sn][r] + bv;
        if (relu) v = fmaxf(v, 0.f);
        C[(size_t)row * N + col] = v;
      }
    }
  }
}

// ---------------------------------------------------------------------------
// Fused masked attention (flash style). Q,K,V: [B, S, H*64] f32 (Q pre-scale
// applied here). One block = (b, head, 64 query rows); 4 waves x 16 rows.
// Streams keys in 32-wide tiles; multi-scale window + pad mask; online
// softmax with cross-lane reductions; PV accumulated with WMMA.
// ---------------------------------------------------------------------------
__global__ __launch_bounds__(128) void attn_fused_kernel(
    const float* __restrict__ Q, const float* __restrict__ Km,
    const float* __restrict__ Vm, const int* __restrict__ pad_num,
    float* __restrict__ O)
{
  __shared__ __align__(16) __bf16 Ks[32][72];      // [key][d]
  __shared__ __align__(16) __bf16 Vs[64][40];      // transposed: [d][key]
  __shared__ __align__(16) __bf16 Ps[4][16][40];   // per-wave P tile [q][key]

  const int tid  = threadIdx.x;
  const int lane = tid & 31;
  const int ln   = lane & 15;
  const int lh   = lane >> 4;
  const int wave = tid >> 5;

  const int nqb  = SS / 64; // 8
  const int qblk = blockIdx.x % nqb;
  const int head = (blockIdx.x / nqb) % HH;
  const int b    = blockIdx.x / (nqb * HH);

  const int q0    = qblk * 64 + wave * 16;
  const int limit = SS - pad_num[b];

  // Q fragments for this wave's 16 rows, pre-scaled by 1/sqrt(64) = 0.125.
  const int qrow = q0 + ln;
  const float* qg = Q + ((size_t)(b * SS + qrow)) * DD + head * DH;
  v16bf qf[2];
  #pragma unroll
  for (int f = 0; f < 2; ++f) {
    float4 a0 = *(const float4*)(qg + f * 32 + lh * 8);
    float4 a1 = *(const float4*)(qg + f * 32 + lh * 8 + 4);
    float4 a2 = *(const float4*)(qg + f * 32 + 16 + lh * 8);
    float4 a3 = *(const float4*)(qg + f * 32 + 16 + lh * 8 + 4);
    v16bf t;
    t[0]  = (__bf16)(a0.x * 0.125f); t[1]  = (__bf16)(a0.y * 0.125f);
    t[2]  = (__bf16)(a0.z * 0.125f); t[3]  = (__bf16)(a0.w * 0.125f);
    t[4]  = (__bf16)(a1.x * 0.125f); t[5]  = (__bf16)(a1.y * 0.125f);
    t[6]  = (__bf16)(a1.z * 0.125f); t[7]  = (__bf16)(a1.w * 0.125f);
    t[8]  = (__bf16)(a2.x * 0.125f); t[9]  = (__bf16)(a2.y * 0.125f);
    t[10] = (__bf16)(a2.z * 0.125f); t[11] = (__bf16)(a2.w * 0.125f);
    t[12] = (__bf16)(a3.x * 0.125f); t[13] = (__bf16)(a3.y * 0.125f);
    t[14] = (__bf16)(a3.z * 0.125f); t[15] = (__bf16)(a3.w * 0.125f);
    qf[f] = t;
  }

  const v8f vzero = {0.f,0.f,0.f,0.f,0.f,0.f,0.f,0.f};
  v8f oacc[4] = {vzero, vzero, vzero, vzero};
  float mrun[8], lrun[8];
  #pragma unroll
  for (int r = 0; r < 8; ++r) { mrun[r] = -1e30f; lrun[r] = 0.f; }

  const bool windowed = head < 6;
  const int step = windowed ? (4 << (head >> 1)) : 0;
  const int s2   = step >> 1;

  for (int k0 = 0; k0 < SS; k0 += 32) {
    // Stage K tile 32x64: float4 global loads -> packed bf16 ds_store_b64.
    #pragma unroll
    for (int i = 0; i < 4; ++i) {
      int chunk = i * 128 + tid;
      int kk = chunk >> 4, d4 = (chunk & 15) * 4;
      float4 kv = *(const float4*)&Km[((size_t)(b * SS + k0 + kk)) * DD +
                                      head * DH + d4];
      uint2 p;
      p.x = pack_bf16(kv.x, kv.y);
      p.y = pack_bf16(kv.z, kv.w);
      *(uint2*)&Ks[kk][d4] = p;
    }
    // Stage V tile transposed [d][key]: pack key-pairs -> ds_store_b32.
    #pragma unroll
    for (int i = 0; i < 8; ++i) {
      int pidx = i * 128 + tid;
      int k2 = pidx >> 6, d = pidx & 63;
      const float* vp = &Vm[((size_t)(b * SS + k0 + 2 * k2)) * DD +
                            head * DH + d];
      *(unsigned*)&Vs[d][2 * k2] = pack_bf16(vp[0], vp[DD]);
    }
    // Prefetch next key-tile of K and V (global_prefetch_b8).
    if (k0 + 32 < SS) {
      int kk = tid >> 2, d16 = (tid & 3) * 16;
      size_t g = ((size_t)(b * SS + k0 + 32 + kk)) * DD + head * DH + d16;
      __builtin_prefetch(&Km[g], 0, 3);
      __builtin_prefetch(&Vm[g], 0, 3);
    }
    __syncthreads();

    // Scores: two 16x16 tiles (keys t*16..t*16+15), K-dim 64 = 2 WMMAs.
    float sc[2][8];
    #pragma unroll
    for (int t = 0; t < 2; ++t) {
      const __bf16* brow = &Ks[t * 16 + ln][0];
      v16bf b0 = ld_frag(brow + lh * 8,      brow + 16 + lh * 8);
      v16bf b1 = ld_frag(brow + 32 + lh * 8, brow + 48 + lh * 8);
      v8f s = vzero;
      s = wmma_bf16(qf[0], b0, s);
      s = wmma_bf16(qf[1], b1, s);
      int key = k0 + t * 16 + ln;
      #pragma unroll
      for (int r = 0; r < 8; ++r) {
        int qi = q0 + r + 8 * lh;
        bool ok = key < limit;
        if (windowed) {
          int left  = (qi < SS / 2) ? imin(qi, s2) : (step - imin(SS - 1 - qi, s2));
          int right = step - left;
          ok = ok && (key >= qi - left) && (key <= qi + right);
        }
        sc[t][r] = ok ? s[r] : -1e9f;
      }
    }

    // Online softmax; row r lives in lanes {lh half}, reduce across 16 lanes.
    #pragma unroll
    for (int r = 0; r < 8; ++r) {
      float tm = fmaxf(sc[0][r], sc[1][r]);
      #pragma unroll
      for (int m = 1; m <= 8; m <<= 1) tm = fmaxf(tm, __shfl_xor(tm, m, 32));
      float mnew  = fmaxf(mrun[r], tm);
      float alpha = expf(mrun[r] - mnew);
      float p0 = expf(sc[0][r] - mnew);
      float p1 = expf(sc[1][r] - mnew);
      float ps = p0 + p1;
      #pragma unroll
      for (int m = 1; m <= 8; m <<= 1) ps += __shfl_xor(ps, m, 32);
      lrun[r] = lrun[r] * alpha + ps;
      mrun[r] = mnew;
      #pragma unroll
      for (int g = 0; g < 4; ++g) oacc[g][r] *= alpha;
      Ps[wave][r + 8 * lh][ln]      = (__bf16)p0;
      Ps[wave][r + 8 * lh][16 + ln] = (__bf16)p1;
    }

    // PV: A = P (wave-private LDS, in-order DS pipe => no barrier needed),
    // B = V tile (transposed); accumulate 16x64 output in 4 WMMAs.
    const __bf16* prow = &Ps[wave][ln][0];
    v16bf pf = ld_frag(prow + lh * 8, prow + 16 + lh * 8);
    #pragma unroll
    for (int g = 0; g < 4; ++g) {
      const __bf16* vrow = &Vs[g * 16 + ln][0];
      v16bf vf = ld_frag(vrow + lh * 8, vrow + 16 + lh * 8);
      oacc[g] = wmma_bf16(pf, vf, oacc[g]);
    }
    __syncthreads();
  }

  #pragma unroll
  for (int g = 0; g < 4; ++g)
    #pragma unroll
    for (int r = 0; r < 8; ++r) {
      int row = q0 + r + 8 * lh;
      O[((size_t)(b * SS + row)) * DD + head * DH + g * 16 + ln] =
          oacc[g][r] / lrun[r];
    }
}

// ---------------------------------------------------------------------------
// out = LayerNorm(T + R) * g + b   (row length 512; one block per row)
// ---------------------------------------------------------------------------
DEV_INLINE float block_sum(float v, float* red) {
  #pragma unroll
  for (int m = 16; m >= 1; m >>= 1) v += __shfl_xor(v, m, 32);
  int w = threadIdx.x >> 5;
  if ((threadIdx.x & 31) == 0) red[w] = v;
  __syncthreads();
  float tot = 0.f;
  #pragma unroll
  for (int i = 0; i < 8; ++i) tot += red[i];
  __syncthreads();
  return tot;
}

__global__ __launch_bounds__(256) void add_ln_kernel(
    const float* __restrict__ T, const float* __restrict__ R,
    const float* __restrict__ g, const float* __restrict__ b,
    float* __restrict__ out)
{
  __shared__ float red[8];
  const int row = blockIdx.x;
  const int tid = threadIdx.x;
  const size_t base = (size_t)row * DD;
  float v0 = T[base + tid]       + R[base + tid];
  float v1 = T[base + tid + 256] + R[base + tid + 256];
  float mu = block_sum(v0 + v1, red) * (1.f / DD);
  float d0 = v0 - mu, d1 = v1 - mu;
  float var = block_sum(d0 * d0 + d1 * d1, red) * (1.f / DD);
  float rstd = rsqrtf(var + 1e-6f);
  out[base + tid]       = d0 * rstd * g[tid]       + b[tid];
  out[base + tid + 256] = d1 * rstd * g[tid + 256] + b[tid + 256];
}

// ---------------------------------------------------------------------------
// Host orchestration (graph-capture safe: kernel launches only).
// ---------------------------------------------------------------------------
extern "C" void kernel_launch(void* const* d_in, const int* in_sizes, int n_in,
                              void* d_out, int out_size, void* d_ws, size_t ws_size,
                              hipStream_t stream)
{
  (void)in_sizes; (void)n_in; (void)out_size; (void)ws_size;
  const float* src      = (const float*)d_in[0];
  const float* trg      = (const float*)d_in[1];
  const int*   enc_pad  = (const int*)d_in[2];
  const int*   dec_pad  = (const int*)d_in[3];
  const float* e_wq  = (const float*)d_in[4];
  const float* e_wk  = (const float*)d_in[5];
  const float* e_wv  = (const float*)d_in[6];
  const float* e_wo  = (const float*)d_in[7];
  const float* e_g1  = (const float*)d_in[8];
  const float* e_b1  = (const float*)d_in[9];
  const float* e_w1  = (const float*)d_in[10];
  const float* e_bb1 = (const float*)d_in[11];
  const float* e_w2  = (const float*)d_in[12];
  const float* e_bb2 = (const float*)d_in[13];
  const float* e_g2  = (const float*)d_in[14];
  const float* e_b2  = (const float*)d_in[15];
  const float* d_swq = (const float*)d_in[16];
  const float* d_swk = (const float*)d_in[17];
  const float* d_swv = (const float*)d_in[18];
  const float* d_swo = (const float*)d_in[19];
  const float* d_g1  = (const float*)d_in[20];
  const float* d_b1  = (const float*)d_in[21];
  const float* d_cwq = (const float*)d_in[22];
  const float* d_cwk = (const float*)d_in[23];
  const float* d_cwv = (const float*)d_in[24];
  const float* d_cwo = (const float*)d_in[25];
  const float* d_g2  = (const float*)d_in[26];
  const float* d_b2  = (const float*)d_in[27];
  const float* d_w1  = (const float*)d_in[28];
  const float* d_bb1 = (const float*)d_in[29];
  const float* d_w2  = (const float*)d_in[30];
  const float* d_bb2 = (const float*)d_in[31];
  const float* d_g3  = (const float*)d_in[32];
  const float* d_b3  = (const float*)d_in[33];

  const size_t SZ_X = (size_t)NTOK * DD;   // 2,097,152 floats
  float* ws = (float*)d_ws;
  float* X  = ws;            // encoder state
  float* Y  = X  + SZ_X;     // decoder state
  float* Qb = Y  + SZ_X;
  float* Kb = Qb + SZ_X;
  float* Vb = Kb + SZ_X;
  float* Ob = Vb + SZ_X;     // attention output
  float* Tb = Ob + SZ_X;     // gemm temp
  float* Hb = Tb + SZ_X;     // FFN hidden (NTOK*DFF floats)

  dim3 blk256(256), blk128(128);

  auto gemm = [&](const float* A, const float* W, const float* bias,
                  float* C, int N, int K, int relu) {
    dim3 grid(N / GBN, NTOK / GBM);
    gemm_bf16_kernel<<<grid, blk256, 0, stream>>>(A, W, bias, C, N, K, relu);
  };
  auto attn = [&](const float* q, const float* k, const float* v,
                  const int* pad, float* o) {
    attn_fused_kernel<<<dim3(BB * HH * (SS / 64)), blk128, 0, stream>>>(
        q, k, v, pad, o);
  };
  auto addln = [&](const float* T, const float* R, const float* g,
                   const float* b, float* o) {
    add_ln_kernel<<<dim3(NTOK), blk256, 0, stream>>>(T, R, g, b, o);
  };

  // ----- Encoder -----
  add_pe_kernel<<<dim3((unsigned)(SZ_X / 256)), blk256, 0, stream>>>(src, X);
  for (int l = 0; l < NL; ++l) {
    size_t wo = (size_t)l * DD * DD;
    gemm(X, e_wq + wo, nullptr, Qb, DD, DD, 0);
    gemm(X, e_wk + wo, nullptr, Kb, DD, DD, 0);
    gemm(X, e_wv + wo, nullptr, Vb, DD, DD, 0);
    attn(Qb, Kb, Vb, enc_pad, Ob);
    gemm(Ob, e_wo + wo, nullptr, Tb, DD, DD, 0);
    addln(Tb, X, e_g1 + l * DD, e_b1 + l * DD, X);
    gemm(X, e_w1 + (size_t)l * DD * DFF, e_bb1 + l * DFF, Hb, DFF, DD, 1);
    gemm(Hb, e_w2 + (size_t)l * DFF * DD, e_bb2 + l * DD, Tb, DD, DFF, 0);
    addln(Tb, X, e_g2 + l * DD, e_b2 + l * DD, X);
  }

  // ----- Decoder -----
  add_pe_kernel<<<dim3((unsigned)(SZ_X / 256)), blk256, 0, stream>>>(trg, Y);
  for (int l = 0; l < NL; ++l) {
    size_t wo = (size_t)l * DD * DD;
    // self attention
    gemm(Y, d_swq + wo, nullptr, Qb, DD, DD, 0);
    gemm(Y, d_swk + wo, nullptr, Kb, DD, DD, 0);
    gemm(Y, d_swv + wo, nullptr, Vb, DD, DD, 0);
    attn(Qb, Kb, Vb, dec_pad, Ob);
    gemm(Ob, d_swo + wo, nullptr, Tb, DD, DD, 0);
    addln(Tb, Y, d_g1 + l * DD, d_b1 + l * DD, Y);
    // cross attention (K/V from encoder output X, src mask)
    gemm(Y, d_cwq + wo, nullptr, Qb, DD, DD, 0);
    gemm(X, d_cwk + wo, nullptr, Kb, DD, DD, 0);
    gemm(X, d_cwv + wo, nullptr, Vb, DD, DD, 0);
    attn(Qb, Kb, Vb, enc_pad, Ob);
    gemm(Ob, d_cwo + wo, nullptr, Tb, DD, DD, 0);
    addln(Tb, Y, d_g2 + l * DD, d_b2 + l * DD, Y);
    // FFN
    gemm(Y, d_w1 + (size_t)l * DD * DFF, d_bb1 + l * DFF, Hb, DFF, DD, 1);
    gemm(Hb, d_w2 + (size_t)l * DFF * DD, d_bb2 + l * DD, Tb, DD, DFF, 0);
    addln(Tb, Y, d_g3 + l * DD, d_b3 + l * DD,
          (l == NL - 1) ? (float*)d_out : Y);
  }
}
